// ResidualVectorQuantizer_85194971283976
// MI455X (gfx1250) — matrix operand order repaired and checked by reading, not compiled, not run on hardware
//
#include <hip/hip_runtime.h>

typedef __attribute__((ext_vector_type(16))) __bf16 bf16x16;
typedef __attribute__((ext_vector_type(8)))  __bf16 bf16x8;
typedef __attribute__((ext_vector_type(4)))  __bf16 bf16x4;
typedef __attribute__((ext_vector_type(8)))  float  f32x8;

#define B_    8
#define D_    256
#define T_    4096
#define NQ    8
#define BINS  1024
#define MROWS (B_ * T_)          // 32768 rows of [B,T,D]

// workspace byte offsets
#define WS_RESF32   0                         // 32768*256*4  = 33554432
#define WS_RESBF16  (33554432)                // 32768*256*2  = 16777216
#define WS_CBBF16   (WS_RESBF16 + 16777216)   // 8*1024*256*2 =  4194304
#define WS_NORMS    (WS_CBBF16  +  4194304)   // 8*1024*4     =    32768
#define WS_LOSSPART (WS_NORMS   +    32768)   // 8*128*4      =     4096

// d_out float-element offsets: quantized | codes | bw | commit_loss
#define OUT_CODES_OFF 8388608                  // B_*D_*T_
#define OUT_SCAL_OFF  8650752                  // + NQ*B_*T_

#define LDS_PITCH 264                          // halves per code row (bank spread)

// ---------------------------------------------------------------------------
// CDNA5 async global->LDS copy (ASYNCcnt-tracked), inline asm per ISA 15.18.3.
// vdst = per-lane LDS byte address, vaddr = per-lane 64-bit global address.
// ---------------------------------------------------------------------------
__device__ __forceinline__ void async_ld_b128(unsigned ldsByteAddr, const void* g) {
    unsigned long long ga = (unsigned long long)(uintptr_t)g;
    asm volatile("global_load_async_to_lds_b128 %0, %1, off"
                 :: "v"(ldsByteAddr), "v"(ga) : "memory");
}
__device__ __forceinline__ void wait_async0() {
    asm volatile("s_wait_asynccnt 0x0" ::: "memory");
}
__device__ __forceinline__ void wait_async2() {
    asm volatile("s_wait_asynccnt 0x2" ::: "memory");
}

// ---------------------------------------------------------------------------
// Codebook prep: fp32 -> bf16 copy + 0.5*||cb_k||^2 per code row.
// ---------------------------------------------------------------------------
__global__ __launch_bounds__(256) void prep_cb(const float* __restrict__ cb,
                                               __bf16* __restrict__ cbh,
                                               float* __restrict__ norms) {
    int rowid = blockIdx.x;                 // i*BINS + k  (8192 rows)
    int d     = threadIdx.x;
    float v = cb[(size_t)rowid * D_ + d];
    cbh[(size_t)rowid * D_ + d] = (__bf16)v;
    float sq = v * v;
#pragma unroll
    for (int off = 16; off >= 1; off >>= 1) sq += __shfl_xor(sq, off, 32);
    __shared__ float red[8];
    int lane = threadIdx.x & 31, wv = threadIdx.x >> 5;
    if (lane == 0) red[wv] = sq;
    __syncthreads();
    if (threadIdx.x == 0) {
        float s = 0.f;
#pragma unroll
        for (int w = 0; w < 8; ++w) s += red[w];
        norms[rowid] = 0.5f * s;
    }
}

// ---------------------------------------------------------------------------
// x[B,D,T] -> residual[B,T,D] (f32 + bf16), tiled LDS transpose.
// grid (T/32, D/32, B), block (32,8)
// ---------------------------------------------------------------------------
__global__ __launch_bounds__(256) void transpose_in(const float* __restrict__ x,
                                                    float* __restrict__ resF32,
                                                    __bf16* __restrict__ resBF16) {
    __shared__ float tile[32][33];
    int b = blockIdx.z, d0 = blockIdx.y * 32, t0 = blockIdx.x * 32;
    int tx = threadIdx.x, ty = threadIdx.y;
#pragma unroll
    for (int j = 0; j < 4; ++j) {
        int d = d0 + ty + j * 8;
        tile[ty + j * 8][tx] = x[((size_t)(b * D_ + d)) * T_ + t0 + tx];
    }
    __syncthreads();
#pragma unroll
    for (int j = 0; j < 4; ++j) {
        int t = t0 + ty + j * 8;
        float v = tile[tx][ty + j * 8];
        size_t o = ((size_t)(b * T_ + t)) * D_ + d0 + tx;
        resF32[o]  = v;
        resBF16[o] = (__bf16)v;
    }
}

// ---------------------------------------------------------------------------
// Fused 8-stage RVQ. Block = 256 rows (8 waves x 32 rows: two 16-row M-tiles
// per wave so each LDS B-load feeds two WMMAs). Codebook tiles double-buffered
// in LDS via async global->LDS loads (ASYNCcnt) prefetched one tile ahead.
// ---------------------------------------------------------------------------
__global__ __launch_bounds__(256) void rvq_main(const float* __restrict__ cbF32,
                                                const __bf16* __restrict__ cbBF16,
                                                const float* __restrict__ norms,
                                                float* __restrict__ resF32,
                                                __bf16* __restrict__ resBF16,
                                                float* __restrict__ outCodes,
                                                float* __restrict__ lossPart) {
    __shared__ __bf16 ldsB[2][16 * LDS_PITCH];   // 2 x (16 codes x 256 K)
    __shared__ float wsum[8];

    const int tid   = threadIdx.x;
    const int lane  = tid & 31;
    const int wv    = tid >> 5;
    const int halfk = lane >> 4;          // 0|1 : which K-half this lane holds
    const int lmod  = lane & 15;
    const int mBase = blockIdx.x * 256 + wv * 32;

    // this thread's staging slice: 32 B of the 8 KB tile
    const int sc  = tid >> 4;             // code row 0..15
    const int skb = (tid & 15) * 16;      // K base (halves)
    const unsigned ldsDst0 = (unsigned)(uintptr_t)&ldsB[0][sc * LDS_PITCH + skb];
    const unsigned ldsDst1 = (unsigned)(uintptr_t)&ldsB[1][sc * LDS_PITCH + skb];

    for (int i = 0; i < NQ; ++i) {
        const __bf16* cbh = cbBF16 + (size_t)i * BINS * D_;
        const float*  cbf = cbF32  + (size_t)i * BINS * D_;
        const float*  nrm = norms  + i * BINS;

        __syncthreads();                   // LDS buffers free (prev stage done)
        {   // prologue: async-stage tile 0 into buffer 0
            const __bf16* src = cbh + (size_t)sc * D_ + skb;
            async_ld_b128(ldsDst0,      src);
            async_ld_b128(ldsDst0 + 16, src + 8);
        }

        // ---- preload two A tiles (32 rows x K=256) in WMMA register layout --
        bf16x16 a0[8], a1[8];
        const __bf16* ar0 = resBF16 + (size_t)(mBase + lmod) * D_ + halfk * 8;
        const __bf16* ar1 = ar0 + (size_t)16 * D_;
#pragma unroll
        for (int k = 0; k < 8; ++k) {
            union { bf16x16 v; bf16x8 h[2]; } u;
            u.h[0] = *(const bf16x8*)(ar0 + k * 32);
            u.h[1] = *(const bf16x8*)(ar0 + k * 32 + 16);
            a0[k] = u.v;
            u.h[0] = *(const bf16x8*)(ar1 + k * 32);
            u.h[1] = *(const bf16x8*)(ar1 + k * 32 + 16);
            a1[k] = u.v;
        }

        float bVal0[8], bVal1[8];
        int   bIdx0[8], bIdx1[8];
#pragma unroll
        for (int r = 0; r < 8; ++r) {
            bVal0[r] = -3.4e38f; bIdx0[r] = 0;
            bVal1[r] = -3.4e38f; bIdx1[r] = 0;
        }

        // ---- sweep 64 tiles of 16 codes, double-buffered async prefetch ----
        for (int j = 0; j < 64; ++j) {
            if (j + 1 < 64) {              // prefetch tile j+1 into other buffer
                const __bf16* src = cbh + (size_t)((j + 1) * 16 + sc) * D_ + skb;
                unsigned dst = ((j + 1) & 1) ? ldsDst1 : ldsDst0;
                async_ld_b128(dst,      src);
                async_ld_b128(dst + 16, src + 8);
                wait_async2();             // own tile-j loads retired
            } else {
                wait_async0();
            }
            __syncthreads();               // every wave's tile-j data resident

            const __bf16* bufj = &ldsB[j & 1][0];
            f32x8 c0 = {0.f, 0.f, 0.f, 0.f, 0.f, 0.f, 0.f, 0.f};
            f32x8 c1 = {0.f, 0.f, 0.f, 0.f, 0.f, 0.f, 0.f, 0.f};
#pragma unroll
            for (int k = 0; k < 8; ++k) {
                const __bf16* bp = bufj + lmod * LDS_PITCH + halfk * 8 + k * 32;
                union { bf16x16 v; bf16x8 h[2]; } ub;
                ub.h[0] = *(const bf16x8*)(bp);
                ub.h[1] = *(const bf16x8*)(bp + 16);
                c0 = __builtin_amdgcn_wmma_f32_16x16x32_bf16(
                        false, a0[k], false, ub.v, (short)0, c0, false, false);
                c1 = __builtin_amdgcn_wmma_f32_16x16x32_bf16(
                        false, a1[k], false, ub.v, (short)0, c1, false, false);
            }
            __syncthreads();               // buffer j&1 free for prefetch j+2

            float bias = nrm[j * 16 + lmod];
            int   code = j * 16 + lmod;
#pragma unroll
            for (int r = 0; r < 8; ++r) {
                float s0 = c0[r] - bias;   // lanes<16: M=r ; lanes>=16: M=r+8
                if (s0 > bVal0[r]) { bVal0[r] = s0; bIdx0[r] = code; }
                float s1 = c1[r] - bias;
                if (s1 > bVal1[r]) { bVal1[r] = s1; bIdx1[r] = code; }
            }
        }

        // ---- argmax butterfly within each 16-lane half ----
#pragma unroll
        for (int r = 0; r < 8; ++r) {
            float v0 = bVal0[r]; int i0 = bIdx0[r];
            float v1 = bVal1[r]; int i1 = bIdx1[r];
#pragma unroll
            for (int off = 1; off < 16; off <<= 1) {
                float w0 = __shfl_xor(v0, off, 32); int k0 = __shfl_xor(i0, off, 32);
                if (w0 > v0 || (w0 == v0 && k0 < i0)) { v0 = w0; i0 = k0; }
                float w1 = __shfl_xor(v1, off, 32); int k1 = __shfl_xor(i1, off, 32);
                if (w1 > v1 || (w1 == v1 && k1 < i1)) { v1 = w1; i1 = k1; }
            }
            bIdx0[r] = i0;
            bIdx1[r] = i1;
        }

        // ---- gather + residual update + codes + loss (rows wave-private) ----
        float acc = 0.f;
#pragma unroll
        for (int m = 0; m < 32; ++m) {
            int idxm = (m < 16) ? __shfl(bIdx0[m & 7], (m < 8) ? 0 : 16, 32)
                                : __shfl(bIdx1[m & 7], ((m & 15) < 8) ? 0 : 16, 32);
            int row = mBase + m;
            if (lane == m) outCodes[(size_t)i * MROWS + row] = (float)idxm;
            const float4* qrow = (const float4*)(cbf + (size_t)idxm * D_);
            float4* rrow = (float4*)(resF32 + (size_t)row * D_);
            __bf16* hrow = resBF16 + (size_t)row * D_;
#pragma unroll
            for (int v = 0; v < 2; ++v) {
                int c4 = v * 32 + lane;
                float4 q = qrow[c4];
                float4 r = rrow[c4];
                r.x -= q.x; r.y -= q.y; r.z -= q.z; r.w -= q.w;
                rrow[c4] = r;
                bf16x4 h;
                h[0] = (__bf16)r.x; h[1] = (__bf16)r.y;
                h[2] = (__bf16)r.z; h[3] = (__bf16)r.w;
                *(bf16x4*)(hrow + c4 * 4) = h;
                acc += r.x * r.x + r.y * r.y + r.z * r.z + r.w * r.w;
            }
        }
#pragma unroll
        for (int off = 16; off >= 1; off >>= 1) acc += __shfl_xor(acc, off, 32);
        __syncthreads();
        if (lane == 0) wsum[wv] = acc;
        __syncthreads();
        if (tid == 0) {
            float s = 0.f;
#pragma unroll
            for (int w = 0; w < 8; ++w) s += wsum[w];
            lossPart[i * 128 + blockIdx.x] = s;   // deterministic: no atomics
        }
    }
}

// ---------------------------------------------------------------------------
// quantized[B,D,T] = x - residual_final (transpose read of residual).
// ---------------------------------------------------------------------------
__global__ __launch_bounds__(256) void finalize_q(const float* __restrict__ x,
                                                  const float* __restrict__ resF32,
                                                  float* __restrict__ outQ) {
    __shared__ float tile[32][33];
    int b = blockIdx.z, d0 = blockIdx.y * 32, t0 = blockIdx.x * 32;
    int tx = threadIdx.x, ty = threadIdx.y;
#pragma unroll
    for (int j = 0; j < 4; ++j) {
        int t = t0 + ty + j * 8;
        tile[ty + j * 8][tx] = resF32[((size_t)(b * T_ + t)) * D_ + d0 + tx];
    }
    __syncthreads();
#pragma unroll
    for (int j = 0; j < 4; ++j) {
        int d = d0 + ty + j * 8;
        size_t o = ((size_t)(b * D_ + d)) * T_ + t0 + tx;
        outQ[o] = x[o] - tile[tx][ty + j * 8];
    }
}

// ---------------------------------------------------------------------------
// bandwidth scalar + commit loss (mean over stages of 0.25*mean(residual^2)).
// lossPart has 8*128 entries.
// ---------------------------------------------------------------------------
__global__ __launch_bounds__(256) void finalize_scalars(const float* __restrict__ lossPart,
                                                        const int* __restrict__ srp,
                                                        float* __restrict__ outSc) {
    float s = 0.f;
#pragma unroll
    for (int k = 0; k < 4; ++k) s += lossPart[threadIdx.x + k * 256];
#pragma unroll
    for (int off = 16; off >= 1; off >>= 1) s += __shfl_xor(s, off, 32);
    __shared__ float red[8];
    int lane = threadIdx.x & 31, wv = threadIdx.x >> 5;
    if (lane == 0) red[wv] = s;
    __syncthreads();
    if (threadIdx.x == 0) {
        float tot = 0.f;
#pragma unroll
        for (int w = 0; w < 8; ++w) tot += red[w];
        float commit = tot * (0.25f / ((float)MROWS * (float)D_ * (float)NQ));
        float bw = (float)(NQ * 10 * srp[0]) / 1000.0f;   // log2(1024)=10
        outSc[0] = bw;
        outSc[1] = commit;
    }
}

// ---------------------------------------------------------------------------
extern "C" void kernel_launch(void* const* d_in, const int* in_sizes, int n_in,
                              void* d_out, int out_size, void* d_ws, size_t ws_size,
                              hipStream_t stream) {
    (void)in_sizes; (void)n_in; (void)out_size; (void)ws_size;
    const float* x  = (const float*)d_in[0];
    const float* cb = (const float*)d_in[1];
    const int*   sr = (const int*)d_in[2];
    float* out = (float*)d_out;

    char* ws = (char*)d_ws;
    float*  resF32   = (float*)(ws + WS_RESF32);
    __bf16* resBF16  = (__bf16*)(ws + WS_RESBF16);
    __bf16* cbBF16   = (__bf16*)(ws + WS_CBBF16);
    float*  norms    = (float*)(ws + WS_NORMS);
    float*  lossPart = (float*)(ws + WS_LOSSPART);

    prep_cb<<<NQ * BINS, 256, 0, stream>>>(cb, cbBF16, norms);
    transpose_in<<<dim3(T_ / 32, D_ / 32, B_), dim3(32, 8), 0, stream>>>(x, resF32, resBF16);
    rvq_main<<<MROWS / 256, 256, 0, stream>>>(cb, cbBF16, norms, resF32, resBF16,
                                              out + OUT_CODES_OFF, lossPart);
    finalize_q<<<dim3(T_ / 32, D_ / 32, B_), dim3(32, 8), 0, stream>>>(x, resF32, out);
    finalize_scalars<<<1, 256, 0, stream>>>(lossPart, sr, out + OUT_SCAL_OFF);
}
